// GNNLayer_50800873177111
// MI455X (gfx1250) — matrix operand order repaired and checked by reading, not compiled, not run on hardware
//
#include <hip/hip_runtime.h>

// ---------------------------------------------------------------------------
// GNN MetaLayer for MI455X (gfx1250): WMMA f16 GEMMs, two-pass BatchNorm,
// atomic scatter-mean aggregation. Edge/node MLP matrix math goes through
// v_wmma_f32_16x16x32_f16 (wave32, 16x16 tiles).
// ---------------------------------------------------------------------------

typedef __attribute__((ext_vector_type(16))) _Float16 v16h;
typedef __attribute__((ext_vector_type(8)))  float    v8f;

#define NNODES  100000
#define NEDGES  1600000
#define NGRAPHS 512
#define ETILES  (NEDGES / 16)
#define NTILES  (NNODES / 16)

union Frag16 { int4 q[2]; v16h h; };

// Per-wave LDS ordering: DS ops are in-order per wave on CDNA5; the wait +
// memory clobber keeps the compiler from reordering across the stage point.
__device__ __forceinline__ void wave_lds_sync() {
  asm volatile("s_wait_dscnt 0" ::: "memory");
}

// Convert 16 contiguous f32 -> 16 f16 halfs, store as 2x ds_store_b128.
__device__ __forceinline__ void stage16(const float* __restrict__ src, _Float16* dst) {
  union { _Float16 h[16]; int4 q[2]; } tmp;
#pragma unroll
  for (int i = 0; i < 16; i += 4) {
    float4 v = *(const float4*)(src + i);
    tmp.h[i+0] = (_Float16)v.x; tmp.h[i+1] = (_Float16)v.y;
    tmp.h[i+2] = (_Float16)v.z; tmp.h[i+3] = (_Float16)v.w;
  }
  *(int4*)dst = tmp.q[0]; *(int4*)(dst + 8) = tmp.q[1];
}

__device__ __forceinline__ void stage16s(const float* __restrict__ src, float s, _Float16* dst) {
  union { _Float16 h[16]; int4 q[2]; } tmp;
#pragma unroll
  for (int i = 0; i < 16; i += 4) {
    float4 v = *(const float4*)(src + i);
    tmp.h[i+0] = (_Float16)(v.x * s); tmp.h[i+1] = (_Float16)(v.y * s);
    tmp.h[i+2] = (_Float16)(v.z * s); tmp.h[i+3] = (_Float16)(v.w * s);
  }
  *(int4*)dst = tmp.q[0]; *(int4*)(dst + 8) = tmp.q[1];
}

__device__ __forceinline__ void zero32B(_Float16* dst) {
  int4 z; z.x = z.y = z.z = z.w = 0;
  *(int4*)dst = z; *(int4*)(dst + 8) = z;
}

// Fill B-operand fragments into LDS from global f32 weights W[Kact x N]
// (row-major, fan_in x fan_out). Fragment f = n0*KC + k0. Contiguous-k B layout (not ISA table order):
// lane L holds column n = n0*16 + (L&15); its 16 packed halfs are
// k = k0*32 + 16*(L>>4) + h. Rows k >= Kact are zero-padded.
__device__ __forceinline__ void fill_wfrags(_Float16* ldsf, const float* __restrict__ W,
                                            int Kact, int N, int NT, int KC) {
  int total = NT * KC * 32;
  for (int t = threadIdx.x; t < total; t += blockDim.x) {
    int f = t >> 5, L = t & 31;
    int n0 = f / KC, k0 = f - n0 * KC;
    int n = n0 * 16 + (L & 15);
    int kbase = k0 * 32 + 16 * (L >> 4);
    _Float16* dst = ldsf + (size_t)t * 16;
#pragma unroll
    for (int h = 0; h < 16; ++h) {
      int k = kbase + h;
      dst[h] = (k < Kact) ? (_Float16)W[k * N + n] : (_Float16)0.0f;
    }
  }
}

__device__ __forceinline__ v16h load_bfrag(const _Float16* ldsf, int f, int lane) {
  const int4* p = (const int4*)(ldsf + (size_t)(f * 32 + lane) * 16);
  Frag16 u; u.q[0] = p[0]; u.q[1] = p[1];
  return u.h;
}

// A fragment from a row-major 16x128-half LDS stage. Contiguous-k A 16x32 layout (same as B):
// lanes 0-15 hold K {0..15} of row M=lane; lanes 16-31 hold K {16..31}.
__device__ __forceinline__ v16h load_afrag(const _Float16* a_stage, int k0, int lane) {
  int m = lane & 15, grp = lane >> 4;
  const _Float16* base = a_stage + m * 128 + k0 * 32 + grp * 16;
  Frag16 u;
  u.q[0] = *(const int4*)(base);
  u.q[1] = *(const int4*)(base + 8);
  return u.h;
}

// ---------------------------------------------------------------------------
// Edge MLP: e_in = [x[row], x[col], edge_attr, u[batch[row]]]  (K=112 pad 128)
// APPLY=false: accumulate BN sums of h = e_in@W1+b1 (128 ch).
// APPLY=true : recompute h, BN+ReLU, edge_new = a@W2+b2, write + scatter-add.
// ---------------------------------------------------------------------------
template<bool APPLY>
__global__ void __launch_bounds__(256) k_edge_mlp(
    const float* __restrict__ x, const int* __restrict__ eidx,
    const float* __restrict__ ea, const float* __restrict__ u,
    const int* __restrict__ batch,
    const float* __restrict__ w1, const float* __restrict__ b1,
    const float* __restrict__ w2, const float* __restrict__ b2,
    const float* __restrict__ scale, const float* __restrict__ shift,
    float* __restrict__ bn_sum, float* __restrict__ bn_sq,
    float* __restrict__ e_sum_n, float* __restrict__ e_cnt_n,
    float* __restrict__ out_e)
{
  extern __shared__ char smem_raw[];
  _Float16* w1f   = (_Float16*)smem_raw;   // 8*4*32*16 = 16384 halfs
  _Float16* w2f   = w1f + 16384;           // 2*4*32*16 =  4096 halfs
  _Float16* a_all = w2f + 4096;            // 8 waves * 16*128 halfs
  float* lsum = (float*)(a_all + 8 * 2048);
  float* lsq  = lsum + 128;

  fill_wfrags(w1f, w1, 112, 128, 8, 4);
  if (APPLY) fill_wfrags(w2f, w2, 128, 32, 2, 4);
  if (!APPLY) for (int i = threadIdx.x; i < 256; i += blockDim.x) lsum[i] = 0.0f;
  __syncthreads();

  int lane = threadIdx.x & 31, wid = threadIdx.x >> 5;
  int grp = lane >> 4, lc = lane & 15;
  _Float16* a_stage = a_all + wid * 2048;

  float bia[8], sc8[8] = {0}, sh8[8] = {0}, b2v[2] = {0};
#pragma unroll
  for (int n0 = 0; n0 < 8; ++n0) bia[n0] = b1[n0 * 16 + lc];
  if (APPLY) {
#pragma unroll
    for (int n0 = 0; n0 < 8; ++n0) { sc8[n0] = scale[n0*16+lc]; sh8[n0] = shift[n0*16+lc]; }
    b2v[0] = b2[lc]; b2v[1] = b2[16 + lc];
  }
  float ss[8] = {0,0,0,0,0,0,0,0}, qq[8] = {0,0,0,0,0,0,0,0};

  int gwv = blockIdx.x * (blockDim.x >> 5) + wid;
  int nwv = gridDim.x * (blockDim.x >> 5);

  for (int t = gwv; t < ETILES; t += nwv) {
    int ebase = t * 16;
    int e = ebase + lc;
    int r = eidx[e];
    int c = eidx[NEDGES + e];
    _Float16* arow = a_stage + lc * 128;
    stage16(x  + (size_t)r * 32 + grp * 16, arow +  0 + grp * 16);
    stage16(x  + (size_t)c * 32 + grp * 16, arow + 32 + grp * 16);
    stage16(ea + (size_t)e * 32 + grp * 16, arow + 64 + grp * 16);
    if (grp == 0) stage16(u + (size_t)batch[r] * 16, arow + 96);
    else          zero32B(arow + 112);
    wave_lds_sync();

    v16h afr[4];
#pragma unroll
    for (int k0 = 0; k0 < 4; ++k0) afr[k0] = load_afrag(a_stage, k0, lane);

#pragma unroll
    for (int n0 = 0; n0 < 8; ++n0) {
      v8f cc = {};
#pragma unroll
      for (int k0 = 0; k0 < 4; ++k0) {
        v16h bf = load_bfrag(w1f, n0 * 4 + k0, lane);
        cc = __builtin_amdgcn_wmma_f32_16x16x32_f16(false, afr[k0], false, bf,
                                                    (short)0, cc, false, false);
      }
      if (!APPLY) {
#pragma unroll
        for (int rr = 0; rr < 8; ++rr) {
          float v = cc[rr] + bia[n0];
          ss[n0] += v; qq[n0] += v * v;
        }
      } else {
#pragma unroll
        for (int rr = 0; rr < 8; ++rr) {
          float v = fmaxf((cc[rr] + bia[n0]) * sc8[n0] + sh8[n0], 0.0f);
          int m = rr + 8 * grp;
          a_stage[m * 128 + n0 * 16 + lc] = (_Float16)v;   // re-stage as A for GEMM2
        }
      }
    }

    if (APPLY) {
      wave_lds_sync();
      v16h a2[4];
#pragma unroll
      for (int k0 = 0; k0 < 4; ++k0) a2[k0] = load_afrag(a_stage, k0, lane);
#pragma unroll
      for (int n1 = 0; n1 < 2; ++n1) {
        v8f c2 = {};
#pragma unroll
        for (int k0 = 0; k0 < 4; ++k0) {
          v16h bf = load_bfrag(w2f, n1 * 4 + k0, lane);
          c2 = __builtin_amdgcn_wmma_f32_16x16x32_f16(false, a2[k0], false, bf,
                                                      (short)0, c2, false, false);
        }
        int n = n1 * 16 + lc;
#pragma unroll
        for (int rr = 0; rr < 8; ++rr) {
          int ee = ebase + rr + 8 * grp;
          float v = c2[rr] + b2v[n1];
          out_e[(size_t)ee * 32 + n] = v;
          int cn = eidx[NEDGES + ee];
          atomicAdd(&e_sum_n[(size_t)cn * 32 + n], v);
        }
      }
      if (lc == 0) {
#pragma unroll
        for (int rr = 0; rr < 8; ++rr) {
          int ee = ebase + rr + 8 * grp;
          atomicAdd(&e_cnt_n[eidx[NEDGES + ee]], 1.0f);
        }
      }
    }
  }

  if (!APPLY) {
#pragma unroll
    for (int n0 = 0; n0 < 8; ++n0) {
      atomicAdd(&lsum[n0 * 16 + lc], ss[n0]);
      atomicAdd(&lsq [n0 * 16 + lc], qq[n0]);
    }
    __syncthreads();
    for (int i = threadIdx.x; i < 128; i += blockDim.x) {
      atomicAdd(&bn_sum[i], lsum[i]);
      atomicAdd(&bn_sq[i],  lsq[i]);
    }
  }
}

// ---------------------------------------------------------------------------
// Node MLP: n_in = [x, e_aggr_n, u[batch]]  (K=80 pad 128)
// ---------------------------------------------------------------------------
template<bool APPLY>
__global__ void __launch_bounds__(256) k_node_mlp(
    const float* __restrict__ x, const int* __restrict__ batch,
    const float* __restrict__ u,
    const float* __restrict__ e_sum_n, const float* __restrict__ e_cnt_n,
    const float* __restrict__ w1, const float* __restrict__ b1,
    const float* __restrict__ w2, const float* __restrict__ b2,
    const float* __restrict__ scale, const float* __restrict__ shift,
    float* __restrict__ bn_sum, float* __restrict__ bn_sq,
    float* __restrict__ x_sum_g, float* __restrict__ out_x)
{
  extern __shared__ char smem_raw[];
  _Float16* w1f   = (_Float16*)smem_raw;
  _Float16* w2f   = w1f + 16384;
  _Float16* a_all = w2f + 4096;
  float* lsum = (float*)(a_all + 8 * 2048);
  float* lsq  = lsum + 128;

  fill_wfrags(w1f, w1, 80, 128, 8, 4);
  if (APPLY) fill_wfrags(w2f, w2, 128, 32, 2, 4);
  if (!APPLY) for (int i = threadIdx.x; i < 256; i += blockDim.x) lsum[i] = 0.0f;
  __syncthreads();

  int lane = threadIdx.x & 31, wid = threadIdx.x >> 5;
  int grp = lane >> 4, lc = lane & 15;
  _Float16* a_stage = a_all + wid * 2048;

  float bia[8], sc8[8] = {0}, sh8[8] = {0}, b2v[2] = {0};
#pragma unroll
  for (int n0 = 0; n0 < 8; ++n0) bia[n0] = b1[n0 * 16 + lc];
  if (APPLY) {
#pragma unroll
    for (int n0 = 0; n0 < 8; ++n0) { sc8[n0] = scale[n0*16+lc]; sh8[n0] = shift[n0*16+lc]; }
    b2v[0] = b2[lc]; b2v[1] = b2[16 + lc];
  }
  float ss[8] = {0,0,0,0,0,0,0,0}, qq[8] = {0,0,0,0,0,0,0,0};

  int gwv = blockIdx.x * (blockDim.x >> 5) + wid;
  int nwv = gridDim.x * (blockDim.x >> 5);

  for (int t = gwv; t < NTILES; t += nwv) {
    int nbase = t * 16;
    int nd = nbase + lc;
    _Float16* arow = a_stage + lc * 128;
    stage16(x + (size_t)nd * 32 + grp * 16, arow + grp * 16);
    float inv = 1.0f / fmaxf(e_cnt_n[nd], 1.0f);
    stage16s(e_sum_n + (size_t)nd * 32 + grp * 16, inv, arow + 32 + grp * 16);
    if (grp == 0) stage16(u + (size_t)batch[nd] * 16, arow + 64);
    else          zero32B(arow + 80);
    zero32B(arow + 96 + grp * 16);
    wave_lds_sync();

    v16h afr[4];
#pragma unroll
    for (int k0 = 0; k0 < 4; ++k0) afr[k0] = load_afrag(a_stage, k0, lane);

#pragma unroll
    for (int n0 = 0; n0 < 8; ++n0) {
      v8f cc = {};
#pragma unroll
      for (int k0 = 0; k0 < 4; ++k0) {
        v16h bf = load_bfrag(w1f, n0 * 4 + k0, lane);
        cc = __builtin_amdgcn_wmma_f32_16x16x32_f16(false, afr[k0], false, bf,
                                                    (short)0, cc, false, false);
      }
      if (!APPLY) {
#pragma unroll
        for (int rr = 0; rr < 8; ++rr) {
          float v = cc[rr] + bia[n0];
          ss[n0] += v; qq[n0] += v * v;
        }
      } else {
#pragma unroll
        for (int rr = 0; rr < 8; ++rr) {
          float v = fmaxf((cc[rr] + bia[n0]) * sc8[n0] + sh8[n0], 0.0f);
          int m = rr + 8 * grp;
          a_stage[m * 128 + n0 * 16 + lc] = (_Float16)v;
        }
      }
    }

    if (APPLY) {
      wave_lds_sync();
      v16h a2[4];
#pragma unroll
      for (int k0 = 0; k0 < 4; ++k0) a2[k0] = load_afrag(a_stage, k0, lane);
#pragma unroll
      for (int n1 = 0; n1 < 2; ++n1) {
        v8f c2 = {};
#pragma unroll
        for (int k0 = 0; k0 < 4; ++k0) {
          v16h bf = load_bfrag(w2f, n1 * 4 + k0, lane);
          c2 = __builtin_amdgcn_wmma_f32_16x16x32_f16(false, a2[k0], false, bf,
                                                      (short)0, c2, false, false);
        }
        int n = n1 * 16 + lc;
#pragma unroll
        for (int rr = 0; rr < 8; ++rr) {
          int nn = nbase + rr + 8 * grp;
          float v = c2[rr] + b2v[n1];
          out_x[(size_t)nn * 32 + n] = v;
          atomicAdd(&x_sum_g[(size_t)batch[nn] * 32 + n], v);
        }
      }
    }
  }

  if (!APPLY) {
#pragma unroll
    for (int n0 = 0; n0 < 8; ++n0) {
      atomicAdd(&lsum[n0 * 16 + lc], ss[n0]);
      atomicAdd(&lsq [n0 * 16 + lc], qq[n0]);
    }
    __syncthreads();
    for (int i = threadIdx.x; i < 128; i += blockDim.x) {
      atomicAdd(&bn_sum[i], lsum[i]);
      atomicAdd(&bn_sq[i],  lsq[i]);
    }
  }
}

// BN finalize: scale = gamma*rsqrt(var+eps), shift = beta - mu*scale
__global__ void k_bn_finalize(const float* __restrict__ s, const float* __restrict__ q,
                              const float* __restrict__ gamma, const float* __restrict__ beta,
                              float* __restrict__ scale, float* __restrict__ shift,
                              float count, int C) {
  int c = blockIdx.x * blockDim.x + threadIdx.x;
  if (c >= C) return;
  float mu  = s[c] / count;
  float var = q[c] / count - mu * mu;
  float sc  = gamma[c] * rsqrtf(var + 1e-5f);
  scale[c] = sc;
  shift[c] = beta[c] - mu * sc;
}

// Fold per-node edge sums/counts into per-graph sums (batch maps node->graph).
__global__ void k_graph_aggr(const int* __restrict__ batch,
                             const float* __restrict__ e_sum_n, const float* __restrict__ e_cnt_n,
                             float* __restrict__ e_sum_g, float* __restrict__ e_cnt_g,
                             float* __restrict__ n_cnt_g) {
  int t = blockIdx.x * blockDim.x + threadIdx.x;
  if (t >= NNODES * 32) return;
  int nd = t >> 5, ch = t & 31;
  int g = batch[nd];
  atomicAdd(&e_sum_g[(size_t)g * 32 + ch], e_sum_n[(size_t)nd * 32 + ch]);
  if (ch == 0) {
    atomicAdd(&e_cnt_g[g], e_cnt_n[nd]);
    atomicAdd(&n_cnt_g[g], 1.0f);
  }
}

// Global model: 512 rows x [80 -> 16] + BN + ReLU. One block of 512 threads.
__global__ void __launch_bounds__(512) k_global(
    const float* __restrict__ u,
    const float* __restrict__ x_sum_g, const float* __restrict__ n_cnt_g,
    const float* __restrict__ e_sum_g, const float* __restrict__ e_cnt_g,
    const float* __restrict__ gw, const float* __restrict__ gb,
    const float* __restrict__ gg, const float* __restrict__ gbeta,
    float* __restrict__ out_u) {
  __shared__ float s[16], q[16], sc[16], sh[16];
  int g = threadIdx.x;
  if (g < 16) { s[g] = 0.0f; q[g] = 0.0f; }
  __syncthreads();

  float gin[80];
  float ninv = 1.0f / fmaxf(n_cnt_g[g], 1.0f);
  float einv = 1.0f / fmaxf(e_cnt_g[g], 1.0f);
#pragma unroll
  for (int i = 0; i < 16; ++i) gin[i] = u[g * 16 + i];
#pragma unroll
  for (int i = 0; i < 32; ++i) gin[16 + i] = x_sum_g[g * 32 + i] * ninv;
#pragma unroll
  for (int i = 0; i < 32; ++i) gin[48 + i] = e_sum_g[g * 32 + i] * einv;

  float h[16];
#pragma unroll
  for (int j = 0; j < 16; ++j) {
    float acc = gb[j];
    for (int k = 0; k < 80; ++k) acc += gin[k] * gw[k * 16 + j];
    h[j] = acc;
    atomicAdd(&s[j], acc);
    atomicAdd(&q[j], acc * acc);
  }
  __syncthreads();
  if (g < 16) {
    float mu  = s[g] / (float)NGRAPHS;
    float var = q[g] / (float)NGRAPHS - mu * mu;
    float scv = gg[g] * rsqrtf(var + 1e-5f);
    sc[g] = scv; sh[g] = gbeta[g] - mu * scv;
  }
  __syncthreads();
#pragma unroll
  for (int j = 0; j < 16; ++j)
    out_u[g * 16 + j] = fmaxf(h[j] * sc[j] + sh[j], 0.0f);
}

// ---------------------------------------------------------------------------
extern "C" void kernel_launch(void* const* d_in, const int* in_sizes, int n_in,
                              void* d_out, int out_size, void* d_ws, size_t ws_size,
                              hipStream_t stream) {
  const float* x    = (const float*)d_in[0];
  const int*   eidx = (const int*)  d_in[1];
  const float* ea   = (const float*)d_in[2];
  const float* u    = (const float*)d_in[3];
  const int*   batch= (const int*)  d_in[4];
  const float* ew1  = (const float*)d_in[5];
  const float* eb1  = (const float*)d_in[6];
  const float* eg   = (const float*)d_in[7];
  const float* ebeta= (const float*)d_in[8];
  const float* ew2  = (const float*)d_in[9];
  const float* eb2  = (const float*)d_in[10];
  const float* nw1  = (const float*)d_in[11];
  const float* nb1  = (const float*)d_in[12];
  const float* ng   = (const float*)d_in[13];
  const float* nbeta= (const float*)d_in[14];
  const float* nw2  = (const float*)d_in[15];
  const float* nb2  = (const float*)d_in[16];
  const float* gw   = (const float*)d_in[17];
  const float* gb   = (const float*)d_in[18];
  const float* gg   = (const float*)d_in[19];
  const float* gbeta= (const float*)d_in[20];

  float* out_x = (float*)d_out;                       // [100000, 32]
  float* out_e = out_x + (size_t)NNODES * 32;         // [1600000, 32]
  float* out_u = out_e + (size_t)NEDGES * 32;         // [512, 16]

  float* ws      = (float*)d_ws;
  float* e_sum_n = ws;                                // N*32
  float* e_cnt_n = e_sum_n + (size_t)NNODES * 32;     // N
  float* e_sum_g = e_cnt_n + NNODES;                  // G*32
  float* e_cnt_g = e_sum_g + NGRAPHS * 32;            // G
  float* x_sum_g = e_cnt_g + NGRAPHS;                 // G*32
  float* n_cnt_g = x_sum_g + NGRAPHS * 32;            // G
  float* bn_es   = n_cnt_g + NGRAPHS;                 // 128
  float* bn_eq   = bn_es + 128;
  float* bn_ns   = bn_eq + 128;
  float* bn_nq   = bn_ns + 128;
  float* bn_esc  = bn_nq + 128;
  float* bn_esh  = bn_esc + 128;
  float* bn_nsc  = bn_esh + 128;
  float* bn_nsh  = bn_nsc + 128;

  size_t zero_floats = (size_t)((bn_nq + 128) - ws);
  hipMemsetAsync(d_ws, 0, zero_floats * sizeof(float), stream);

  size_t smem = (size_t)(16384 + 4096 + 8 * 2048) * 2 + 256 * sizeof(float);

  k_edge_mlp<false><<<1250, 256, smem, stream>>>(x, eidx, ea, u, batch, ew1, eb1, ew2, eb2,
      bn_esc, bn_esh, bn_es, bn_eq, e_sum_n, e_cnt_n, out_e);
  k_bn_finalize<<<1, 128, 0, stream>>>(bn_es, bn_eq, eg, ebeta, bn_esc, bn_esh,
      (float)NEDGES, 128);
  k_edge_mlp<true><<<1250, 256, smem, stream>>>(x, eidx, ea, u, batch, ew1, eb1, ew2, eb2,
      bn_esc, bn_esh, bn_es, bn_eq, e_sum_n, e_cnt_n, out_e);

  k_graph_aggr<<<(NNODES * 32) / 256, 256, 0, stream>>>(batch, e_sum_n, e_cnt_n,
      e_sum_g, e_cnt_g, n_cnt_g);

  k_node_mlp<false><<<625, 256, smem, stream>>>(x, batch, u, e_sum_n, e_cnt_n,
      nw1, nb1, nw2, nb2, bn_nsc, bn_nsh, bn_ns, bn_nq, x_sum_g, out_x);
  k_bn_finalize<<<1, 128, 0, stream>>>(bn_ns, bn_nq, ng, nbeta, bn_nsc, bn_nsh,
      (float)NNODES, 128);
  k_node_mlp<true><<<625, 256, smem, stream>>>(x, batch, u, e_sum_n, e_cnt_n,
      nw1, nb1, nw2, nb2, bn_nsc, bn_nsh, bn_ns, bn_nq, x_sum_g, out_x);

  k_global<<<1, 512, 0, stream>>>(u, x_sum_g, n_cnt_g, e_sum_g, e_cnt_g,
      gw, gb, gg, gbeta, out_u);
}
